// HybridCausalMambaBlockParallel_44702019617301
// MI455X (gfx1250) — compile-verified, matrix operand order checked by reading, not run on hardware
//
#include <hip/hip_runtime.h>
#include <math.h>

#define DEV __device__ __forceinline__

constexpr int Bn = 4, Ln = 2048, Dn = 256;
constexpr int DIn = 512, DSn = 16;
constexpr int Mrows = Bn * Ln;   // 8192
constexpr int LP = Ln + 2;       // left-padded rows per batch (K=3 causal conv)
constexpr int NC = 16;           // scan chunks
constexpr int CL = Ln / NC;      // chunk length = 128

typedef __attribute__((ext_vector_type(16))) __bf16 v16bf;
typedef __attribute__((ext_vector_type(8)))  float  v8f;

// ---------- helpers ----------
DEV unsigned short f2bf(float f) {            // RNE f32 -> bf16
  unsigned u = __float_as_uint(f);
  u += 0x7FFFu + ((u >> 16) & 1u);
  return (unsigned short)(u >> 16);
}
DEV float sigm(float x) { return 1.f / (1.f + __expf(-x)); }

union FragBF { uint4 u4[2]; v16bf v; };

// ---------- kernel 1: LayerNorm -> bf16 into left-padded buffer ----------
__global__ void ln_pad_kernel(const float* __restrict__ x,
                              const float* __restrict__ w,
                              const float* __restrict__ bia,
                              unsigned short* __restrict__ hpad) {
  int row  = blockIdx.x * (blockDim.x >> 5) + (threadIdx.x >> 5); // one wave per row
  int lane = threadIdx.x & 31;
  if (row >= Mrows) return;
  int bb = row / Ln, l = row - bb * Ln;
  const float* xr = x + (size_t)row * Dn + lane * 8;
  float4 a = *(const float4*)xr;
  float4 c = *(const float4*)(xr + 4);
  float s  = a.x + a.y + a.z + a.w + c.x + c.y + c.z + c.w;
  float s2 = a.x*a.x + a.y*a.y + a.z*a.z + a.w*a.w
           + c.x*c.x + c.y*c.y + c.z*c.z + c.w*c.w;
#pragma unroll
  for (int off = 16; off > 0; off >>= 1) {
    s  += __shfl_xor(s,  off, 32);
    s2 += __shfl_xor(s2, off, 32);
  }
  float mu   = s * (1.f / Dn);
  float rstd = rsqrtf(s2 * (1.f / Dn) - mu * mu + 1e-5f);
  const float* wp = w   + lane * 8;
  const float* bp = bia + lane * 8;
  float o[8] = {a.x, a.y, a.z, a.w, c.x, c.y, c.z, c.w};
  unsigned r[4];
#pragma unroll
  for (int j = 0; j < 4; j++) {
    float lo = (o[2*j]   - mu) * rstd * wp[2*j]   + bp[2*j];
    float hi = (o[2*j+1] - mu) * rstd * wp[2*j+1] + bp[2*j+1];
    r[j] = (unsigned)f2bf(lo) | ((unsigned)f2bf(hi) << 16);
  }
  *(uint4*)(hpad + ((size_t)bb * LP + l + 2) * Dn + lane * 8) =
      make_uint4(r[0], r[1], r[2], r[3]);
  if (l < 2)  // zero the 2 pad rows at the head of each batch
    *(uint4*)(hpad + ((size_t)bb * LP + l) * Dn + lane * 8) = make_uint4(0u,0u,0u,0u);
}

// ---------- kernel 2: weight packing (f32 -> bf16, reorder/pad, A = -exp(A_log)) ----
__global__ void pack_kernel(const float* __restrict__ inW, const float* __restrict__ xpW,
                            const float* __restrict__ dtW, const float* __restrict__ outW,
                            const float* __restrict__ convW, const float* __restrict__ Alog,
                            unsigned short* __restrict__ inWb, unsigned short* __restrict__ xpWb,
                            unsigned short* __restrict__ dtWb, unsigned short* __restrict__ outWb,
                            unsigned short* __restrict__ convWb, float* __restrict__ An) {
  int i = blockIdx.x * blockDim.x + threadIdx.x;
  if (i < 1024 * 256) inWb[i]  = f2bf(inW[i]);                  // (2*DI, D)
  if (i < 64 * 512) {                                           // (48, DI) -> pad N to 64
    int n = i >> 9, k = i & 511;
    xpWb[i] = (n < 48) ? f2bf(xpW[n * 512 + k]) : (unsigned short)0;
  }
  if (i < 512 * 32) {                                           // (DI,16) -> pad K to 32
    int n = i >> 5, k = i & 31;
    dtWb[i] = (k < 16) ? f2bf(dtW[n * 16 + k]) : (unsigned short)0;
  }
  if (i < 256 * 512)  outWb[i] = f2bf(outW[i]);                 // (D, DI)
  if (i < 256 * 768) {                                          // (o,i,tap) -> (o, tap*256+i)
    int o = i / 768, r = i - o * 768;
    int tap = r >> 8, ci = r & 255;
    convWb[i] = f2bf(convW[(o * 256 + ci) * 3 + tap]);
  }
  if (i < 512 * 16)   An[i] = -__expf(Alog[i]);
}

// ---------- kernel 3: WMMA GEMM, wave tile 32(M) x 64(N), K-step 32, no guards ----
// C[m,n] = sum_k A[m,k] * W[n,k]; all shapes compile-time; fused epilogues.
enum { EPI_F32 = 0, EPI_CONV = 1, EPI_XDBL = 2, EPI_DELTA = 3, EPI_OUT = 4 };

template<int EPI, int M, int N, int K, int LDA, int LDW, int LDC, int BPAD>
__global__ void wmma_gemm_kernel(const unsigned short* __restrict__ A,
                                 const unsigned short* __restrict__ W,
                                 float* __restrict__ Cf, unsigned short* __restrict__ Cb,
                                 const float* __restrict__ bias,
                                 const float* __restrict__ xres,
                                 float* __restrict__ outacc) {
  constexpr int MT = 2, NT = 4;                 // 32 x 64 wave tile
  constexpr int tilesN = N / (NT * 16);
  int lane = threadIdx.x & 31;
  int wid  = blockIdx.x * (blockDim.x >> 5) + (threadIdx.x >> 5);
  int tm = wid / tilesN;
  int tn = wid - tm * tilesN;
  if (tm >= M / 32) return;
  int m0 = tm * 32, n0 = tn * 64;
  int r15   = lane & 15;
  int khalf = (lane < 16) ? 0 : 8;   // ISA K-split: lanes 0-15 carry K {0..7,16..23}

  const unsigned short* Ar[MT];
#pragma unroll
  for (int i = 0; i < MT; i++) {
    int arow = m0 + i * 16 + r15;
    Ar[i] = A + (size_t)arow * LDA + (size_t)(arow / Ln) * BPAD + khalf;
  }
  const unsigned short* Wr[NT];
#pragma unroll
  for (int j = 0; j < NT; j++)
    Wr[j] = W + (size_t)(n0 + j * 16 + r15) * LDW + khalf;

  v8f acc[MT][NT];
#pragma unroll
  for (int i = 0; i < MT; i++)
#pragma unroll
    for (int j = 0; j < NT; j++)
      acc[i][j] = v8f{0.f,0.f,0.f,0.f,0.f,0.f,0.f,0.f};

  for (int kk = 0; kk < K; kk += 32) {
    FragBF a[MT], b[NT];
#pragma unroll
    for (int i = 0; i < MT; i++) {
      a[i].u4[0] = *(const uint4*)(Ar[i] + kk);
      a[i].u4[1] = *(const uint4*)(Ar[i] + kk + 16);
    }
#pragma unroll
    for (int j = 0; j < NT; j++) {
      b[j].u4[0] = *(const uint4*)(Wr[j] + kk);
      b[j].u4[1] = *(const uint4*)(Wr[j] + kk + 16);
    }
    if (kk + 32 < K) {                     // prefetch next K-tile
#pragma unroll
      for (int i = 0; i < MT; i++) __builtin_prefetch(Ar[i] + kk + 32, 0, 1);
#pragma unroll
      for (int j = 0; j < NT; j++) __builtin_prefetch(Wr[j] + kk + 32, 0, 1);
    }
#pragma unroll
    for (int i = 0; i < MT; i++)
#pragma unroll
      for (int j = 0; j < NT; j++)
        acc[i][j] = __builtin_amdgcn_wmma_f32_16x16x32_bf16(
            false, a[i].v, false, b[j].v, (short)0, acc[i][j], false, false);
  }

  int mb = ((lane < 16) ? 0 : 8);
#pragma unroll
  for (int i = 0; i < MT; i++) {
#pragma unroll
    for (int j = 0; j < NT; j++) {
      int n = n0 + j * 16 + r15;
#pragma unroll
      for (int r = 0; r < 8; r++) {
        int m = m0 + i * 16 + mb + r;
        float v = acc[i][j][r];
        size_t idx = (size_t)m * LDC + n;
        if (EPI == EPI_F32) {
          Cf[idx] = v;
        } else if (EPI == EPI_CONV) {             // bias + exact GELU + residual mix
          float t = v + bias[n];
          float g = 0.5f * t * (1.f + erff(t * 0.70710678118654752f));
          outacc[idx] = xres[idx] + 0.5f * g;     // x + (1-ALPHA)*h_conv
        } else if (EPI == EPI_XDBL) {             // x_dbl f32 (stride 64) + padded dt bf16
          Cf[idx] = v;
          if (n < 32) Cb[(size_t)m * 32 + n] = (n < 16) ? f2bf(v) : (unsigned short)0;
        } else if (EPI == EPI_DELTA) {            // softplus(dt_proj + b)
          float t = v + bias[n];
          Cf[idx] = (t > 20.f) ? t : log1pf(__expf(t));
        } else {                                  // EPI_OUT: += ALPHA * h_mamba
          outacc[idx] += 0.5f * v;
        }
      }
    }
  }
}

// ---------- kernel 4: depthwise causal conv (DCONV=4) + SiLU ----------
__global__ void dwconv_kernel(const float* __restrict__ xz, const float* __restrict__ w,
                              const float* __restrict__ bias, float* __restrict__ xcf,
                              unsigned short* __restrict__ xcb) {
  int i = blockIdx.x * blockDim.x + threadIdx.x;   // over B*L*DI
  if (i >= Mrows * DIn) return;
  int d = i & (DIn - 1);
  int m = i >> 9;
  int l = m & (Ln - 1);
  float acc = bias[d];
  const float* w4 = w + d * 4;
#pragma unroll
  for (int t = 0; t < 4; t++) {
    int ls = l - 3 + t;
    if (ls >= 0) acc += xz[(size_t)(m - l + ls) * 1024 + d] * w4[t];  // xin = xz[:,:512]
  }
  float s = acc * sigm(acc);
  xcf[i] = s;
  xcb[i] = f2bf(s);
}

// ---------- kernels 5-7: chunked parallel selective scan ----------
// state rec: h_l = exp(delta_l * a) h_{l-1} + delta_l*xc_l*B_l  (per d,s)
// pass 1: per chunk, zero-start state + transition P = exp(a * sum(delta))
__global__ void scan_part1(const float* __restrict__ delta, const float* __restrict__ xcf,
                           const float* __restrict__ xdbl, const float* __restrict__ An,
                           float* __restrict__ chunkP, float* __restrict__ chunkB) {
  __shared__ float sB[16];
  int bb = blockIdx.x >> 4;       // batch
  int ck = blockIdx.x & 15;       // chunk
  int d  = threadIdx.x;           // channel 0..511
  float a[DSn], st[DSn];
#pragma unroll
  for (int s = 0; s < DSn; s++) { a[s] = An[d * DSn + s]; st[s] = 0.f; }
  float sumdl = 0.f;
  size_t base = (size_t)bb * Ln + (size_t)ck * CL;
  for (int l = 0; l < CL; l++) {
    size_t m = base + l;
    float bv = 0.f;
    if (d < 16) bv = xdbl[m * 64 + 16 + d];       // Bm
    __syncthreads();
    if (d < 16) sB[d] = bv;
    float dl = delta[m * DIn + d];
    float xc = xcf[m * DIn + d];
    __syncthreads();
    float dx = dl * xc;
    sumdl += dl;
#pragma unroll
    for (int s = 0; s < DSn; s++) {
      float dA = __expf(dl * a[s]);
      st[s] = dA * st[s] + dx * sB[s];
    }
  }
  size_t o = (((size_t)bb * NC + ck) * DIn + d) * DSn;
#pragma unroll
  for (int s = 0; s < DSn; s++) {
    chunkP[o + s] = __expf(a[s] * sumdl);         // prod of per-step dA, one exp
    chunkB[o + s] = st[s];
  }
}

// pass 2: compose the NC affine maps sequentially per (b,d,s)
__global__ void scan_combine(const float* __restrict__ chunkP, const float* __restrict__ chunkB,
                             float* __restrict__ hstart) {
  int i = blockIdx.x * blockDim.x + threadIdx.x;  // over Bn*DIn*DSn = 32768
  if (i >= Bn * DIn * DSn) return;
  int bb  = i / (DIn * DSn);
  int rem = i - bb * (DIn * DSn);
  float h = 0.f;
#pragma unroll
  for (int c = 0; c < NC; c++) {
    size_t idx = ((size_t)bb * NC + c) * (DIn * DSn) + rem;
    hstart[idx] = h;
    h = chunkP[idx] * h + chunkB[idx];
  }
}

// pass 3: replay chunk from true start state; fuse C-reduce, D-skip, SiLU(z) gate
__global__ void scan_part2(const float* __restrict__ delta, const float* __restrict__ xcf,
                           const float* __restrict__ xz, const float* __restrict__ xdbl,
                           const float* __restrict__ An, const float* __restrict__ Dp,
                           const float* __restrict__ hstart, unsigned short* __restrict__ ybf) {
  __shared__ float sBC[32];
  int bb = blockIdx.x >> 4;
  int ck = blockIdx.x & 15;
  int d  = threadIdx.x;
  float a[DSn], st[DSn];
  size_t hidx = (((size_t)bb * NC + ck) * DIn + d) * DSn;
#pragma unroll
  for (int s = 0; s < DSn; s++) { a[s] = An[d * DSn + s]; st[s] = hstart[hidx + s]; }
  float dp = Dp[d];
  size_t base = (size_t)bb * Ln + (size_t)ck * CL;
  for (int l = 0; l < CL; l++) {
    size_t m = base + l;
    float bc = 0.f;
    if (d < 32) bc = xdbl[m * 64 + 16 + d];       // Bm(16) ++ Cm(16)
    __syncthreads();
    if (d < 32) sBC[d] = bc;
    float dl = delta[m * DIn + d];
    float xc = xcf[m * DIn + d];
    float zv = xz[m * 1024 + DIn + d];            // z half of in_proj
    __syncthreads();
    float y  = 0.f;
    float dx = dl * xc;
#pragma unroll
    for (int s = 0; s < DSn; s++) {
      float dA = __expf(dl * a[s]);               // a[] already negative
      st[s] = dA * st[s] + dx * sBC[s];
      y += st[s] * sBC[16 + s];
    }
    y += xc * dp;
    y *= zv * sigm(zv);                           // * silu(z)
    ybf[m * DIn + d] = f2bf(y);
  }
}

// ---------- host launcher ----------
extern "C" void kernel_launch(void* const* d_in, const int* in_sizes, int n_in,
                              void* d_out, int out_size, void* d_ws, size_t ws_size,
                              hipStream_t stream) {
  (void)in_sizes; (void)n_in; (void)out_size; (void)ws_size;
  const float* x      = (const float*)d_in[0];
  const float* ln_w   = (const float*)d_in[1];
  const float* ln_b   = (const float*)d_in[2];
  const float* conv_w = (const float*)d_in[3];
  const float* conv_b = (const float*)d_in[4];
  const float* inW    = (const float*)d_in[5];
  const float* c1w    = (const float*)d_in[6];
  const float* c1b    = (const float*)d_in[7];
  const float* xpW    = (const float*)d_in[8];
  const float* dtW    = (const float*)d_in[9];
  const float* dtB    = (const float*)d_in[10];
  const float* Alog   = (const float*)d_in[11];
  const float* DpI    = (const float*)d_in[12];
  const float* outW   = (const float*)d_in[13];
  float* out = (float*)d_out;

  char* basep = (char*)d_ws;
  size_t off = 0;
  auto carve = [&](size_t bytes) -> char* {
    char* r = basep + off;
    off += (bytes + 255) & ~(size_t)255;
    return r;
  };
  unsigned short* hpad   = (unsigned short*)carve((size_t)Bn * LP * Dn * 2);
  unsigned short* inWb   = (unsigned short*)carve((size_t)1024 * 256 * 2);
  unsigned short* xpWb   = (unsigned short*)carve((size_t)64 * 512 * 2);
  unsigned short* dtWb   = (unsigned short*)carve((size_t)512 * 32 * 2);
  unsigned short* outWb  = (unsigned short*)carve((size_t)256 * 512 * 2);
  unsigned short* convWb = (unsigned short*)carve((size_t)256 * 768 * 2);
  float*          An     = (float*)carve((size_t)DIn * DSn * 4);
  float*          xzbuf  = (float*)carve((size_t)Mrows * 1024 * 4);
  float*          xcf    = (float*)carve((size_t)Mrows * DIn * 4);
  unsigned short* xcb    = (unsigned short*)carve((size_t)Mrows * DIn * 2);
  float*          xdblf  = (float*)carve((size_t)Mrows * 64 * 4);   // stride-64 x_dbl
  unsigned short* dtb    = (unsigned short*)carve((size_t)Mrows * 32 * 2);
  float*          deltab = (float*)carve((size_t)Mrows * DIn * 4);
  unsigned short* ybf    = (unsigned short*)carve((size_t)Mrows * DIn * 2);
  float*          chunkP = (float*)carve((size_t)Bn * NC * DIn * DSn * 4);
  float*          chunkB = (float*)carve((size_t)Bn * NC * DIn * DSn * 4);
  float*          hstart = (float*)carve((size_t)Bn * NC * DIn * DSn * 4);

  pack_kernel<<<1024, 256, 0, stream>>>(inW, xpW, dtW, outW, conv_w, Alog,
                                        inWb, xpWb, dtWb, outWb, convWb, An);
  ln_pad_kernel<<<Mrows / 8, 256, 0, stream>>>(x, ln_w, ln_b, hpad);

  // conv branch: implicit-im2col GEMM (K=3*256) + bias + GELU + residual -> d_out
  wmma_gemm_kernel<EPI_CONV, Mrows, 256, 768, 256, 768, 256, 512>
      <<<(Mrows/32)*(256/64)/8, 256, 0, stream>>>(hpad, convWb,
          nullptr, nullptr, conv_b, x, out);

  // in_proj: xz = h @ W^T  (M=8192, N=1024, K=256)
  wmma_gemm_kernel<EPI_F32, Mrows, 1024, 256, 256, 256, 1024, 512>
      <<<(Mrows/32)*(1024/64)/8, 256, 0, stream>>>(hpad + 2 * Dn, inWb,
          xzbuf, nullptr, nullptr, nullptr, nullptr);

  dwconv_kernel<<<(Mrows * DIn) / 256, 256, 0, stream>>>(xzbuf, c1w, c1b, xcf, xcb);

  // x_proj: x_dbl = xc @ W^T, N padded 48->64, ldc=64; also emits padded dt bf16
  wmma_gemm_kernel<EPI_XDBL, Mrows, 64, 512, 512, 512, 64, 0>
      <<<(Mrows/32)*(64/64)/8, 256, 0, stream>>>(xcb, xpWb,
          xdblf, dtb, nullptr, nullptr, nullptr);

  // dt_proj (+bias, softplus): K padded 16->32, fully unguarded
  wmma_gemm_kernel<EPI_DELTA, Mrows, 512, 32, 32, 32, 512, 0>
      <<<(Mrows/32)*(512/64)/8, 256, 0, stream>>>(dtb, dtWb,
          deltab, nullptr, dtB, nullptr, nullptr);

  // chunked parallel scan
  scan_part1<<<Bn * NC, DIn, 0, stream>>>(deltab, xcf, xdblf, An, chunkP, chunkB);
  scan_combine<<<(Bn * DIn * DSn) / 256, 256, 0, stream>>>(chunkP, chunkB, hstart);
  scan_part2<<<Bn * NC, DIn, 0, stream>>>(deltab, xcf, xzbuf, xdblf, An, DpI,
                                          hstart, ybf);

  // out_proj: d_out += ALPHA * (y @ W^T)
  wmma_gemm_kernel<EPI_OUT, Mrows, 256, 512, 512, 512, 256, 0>
      <<<(Mrows/32)*(256/64)/8, 256, 0, stream>>>(ybf, outWb,
          nullptr, nullptr, nullptr, nullptr, out);
}
